// DirectionAwareMessagePassingPTS_35862976921771
// MI455X (gfx1250) — matrix operand order repaired and checked by reading, not compiled, not run on hardware
//
#include <hip/hip_runtime.h>
#include <hip/hip_bf16.h>

#define BB 32
#define NN 256
#define RR 2048
#define DD 1024
#define LN_EPS 1e-5f

typedef __attribute__((ext_vector_type(16))) __bf16 v16bf;
typedef __attribute__((ext_vector_type(8)))  __bf16 v8bf;
typedef __attribute__((ext_vector_type(8)))  float  v8f;

union V16 { v16bf v; uint4 q[2]; v8bf h[2]; };

__device__ __forceinline__ v8f wmma_bf16(v16bf a, v16bf b, v8f c) {
  return __builtin_amdgcn_wmma_f32_16x16x32_bf16(false, a, false, b, (short)0, c, false, false);
}

// A fragment (16x32, bf16) from a row-major f32 row. ISA 16-bit A layout:
// lane l holds row M=l&15; element i -> k = (i>=8?16:0) + (l>>4)*8 + (i&7)
__device__ __forceinline__ v16bf frag_a_f32(const float* row, int k0, int lane) {
  const int hh = (lane >> 4) & 1;
  const float* p = row + k0 + hh * 8;
  float4 x0 = *(const float4*)(p);
  float4 x1 = *(const float4*)(p + 4);
  float4 x2 = *(const float4*)(p + 16);
  float4 x3 = *(const float4*)(p + 20);
  v16bf a;
  a[0]=(__bf16)x0.x; a[1]=(__bf16)x0.y; a[2]=(__bf16)x0.z; a[3]=(__bf16)x0.w;
  a[4]=(__bf16)x1.x; a[5]=(__bf16)x1.y; a[6]=(__bf16)x1.z; a[7]=(__bf16)x1.w;
  a[8]=(__bf16)x2.x; a[9]=(__bf16)x2.y; a[10]=(__bf16)x2.z; a[11]=(__bf16)x2.w;
  a[12]=(__bf16)x3.x; a[13]=(__bf16)x3.y; a[14]=(__bf16)x3.z; a[15]=(__bf16)x3.w;
  return a;
}

// Same A fragment from a row-major bf16 row (two 16B runs).
__device__ __forceinline__ v16bf frag_a_bf16(const __bf16* row, int k0, int lane) {
  const int hh = (lane >> 4) & 1;
  const __bf16* p = row + k0 + hh * 8;
  V16 r;
  r.h[0] = *(const v8bf*)(p);
  r.h[1] = *(const v8bf*)(p + 16);
  return r.v;
}

// A fragment from an LDS slot (512 bf16 in (lane,i) fragment order).
__device__ __forceinline__ v16bf frag_slot(const __bf16* lds, int slot, int lane) {
  const __bf16* p = lds + (slot << 9) + (lane << 4);
  V16 r;
  r.h[0] = *(const v8bf*)(p);
  r.h[1] = *(const v8bf*)(p + 8);
  return r.v;
}

// B fragment (32x16) from pre-swizzled weights: tile (nt,kt) stored as 512
// contiguous bf16 in (lane,i) order -> two coalesced 16B loads per lane.
__device__ __forceinline__ v16bf frag_b(const __bf16* Wf, int nt, int kt, int Kt, int lane) {
  const __bf16* p = Wf + (((size_t)nt * Kt + kt) << 9) + (lane << 4);
  V16 r;
  r.q[0] = *(const uint4*)(p);
  r.q[1] = *(const uint4*)(p + 8);
  return r.v;
}

// convert 8 consecutive f32 -> 8 bf16, one 16B store
__device__ __forceinline__ void cvt_store8(__bf16* dst, const float* src) {
  float4 a = *(const float4*)(src);
  float4 b = *(const float4*)(src + 4);
  v8bf o;
  o[0]=(__bf16)a.x; o[1]=(__bf16)a.y; o[2]=(__bf16)a.z; o[3]=(__bf16)a.w;
  o[4]=(__bf16)b.x; o[5]=(__bf16)b.y; o[6]=(__bf16)b.z; o[7]=(__bf16)b.w;
  *(v8bf*)dst = o;
}

// ---------------- weight pre-swizzle (one-time, cheap) ----------------
__global__ void k_swizzle_w(const float* __restrict__ W, __bf16* __restrict__ Wf,
                            int K, int Ncols) {
  size_t idx = (size_t)blockIdx.x * blockDim.x + threadIdx.x;
  if (idx >= (size_t)K * Ncols) return;
  const int Kt = K >> 5;
  int i = idx & 15;
  int lane = (idx >> 4) & 31;
  size_t tile = idx >> 9;
  int kt = (int)(tile % Kt);
  int nt = (int)(tile / Kt);
  int n = nt * 16 + (lane & 15);
  int k = kt * 32 + (i >= 8 ? 16 : 0) + (lane >> 4) * 8 + (i & 7);
  Wf[idx] = (__bf16)W[(size_t)k * Ncols + n];
}

// ------- fused s/o/u GEMM + triple product -> coeff (block = 32 pairs) ------
// Two 16-pair A row-tiles x3 matrices staged in LDS (192 KB, bf16, fragment
// order). LDS slots are interleaved per k-tile (slot = kt*6 + q*2 + tile) so
// all 12 ds_loads of a k-step use immediate offsets from ONE base register.
// Each B weight fragment is shared by two WMMAs -> 6 independent WMMA chains.
__global__ __launch_bounds__(256) void k_coeff(
    const float* __restrict__ obj, const float* __restrict__ uni,
    const __bf16* __restrict__ Wfs, const __bf16* __restrict__ Wfo,
    const __bf16* __restrict__ Wfu,
    const float* __restrict__ ws_b, const float* __restrict__ wo_b,
    const float* __restrict__ wu_b,
    const float* __restrict__ w_w, const float* __restrict__ w_b,
    const int* __restrict__ pairs, float* __restrict__ coeff) {
  extern __shared__ char smem[];
  __bf16* A = (__bf16*)smem;                  // 192 slots x 512 bf16 = 192KB
  float*  red = (float*)(A + 6 * 32768);      // [8][32]

  const int b   = blockIdx.x >> 6;            // RR/32 = 64 pair-groups / batch
  const int pg  = blockIdx.x & 63;
  const int lane = threadIdx.x & 31;
  const int warp = threadIdx.x >> 5;          // 0..7
  const int m = lane & 15;

  // ---- stage 32 A rows x3 (f32 -> bf16, fragment-swizzled slots) ----
  {
    const int ri = threadIdx.x >> 3;          // row 0..31
    const int tile = ri >> 4;                 // 0/1
    const int rowin = ri & 15;
    const int ccol = (threadIdx.x & 7) * 128; // 128-col chunk
    const int prr = pg * 32 + ri;
    const int p0 = pairs[((size_t)b * RR + prr) * 2 + 0];
    const int p1 = pairs[((size_t)b * RR + prr) * 2 + 1];
    const float* sr = obj + ((size_t)b * NN + p0) * DD;
    const float* orr = obj + ((size_t)b * NN + p1) * DD;
    const float* ur = uni + ((size_t)b * RR + prr) * DD;
#pragma unroll
    for (int c = 0; c < 16; ++c) {
      const int k0 = ccol + c * 8;            // aligned run of 8 k
      const int kt = k0 >> 5, kin = k0 & 31;
      const int hh = (kin >> 3) & 1;
      const int i0 = (kin >= 16 ? 8 : 0);
      const int sub = ((hh * 16 + rowin) << 4) + i0;
      const int s6 = kt * 6 + tile;           // q stride = 2 slots
      cvt_store8(A + ((s6 + 0) << 9) + sub, sr + k0);
      cvt_store8(A + ((s6 + 2) << 9) + sub, orr + k0);
      cvt_store8(A + ((s6 + 4) << 9) + sub, ur + k0);
    }
  }
  __syncthreads();

  // ---- 8 N-tiles per wave, K = 1024, 2 A-tiles share each B fragment ----
  v8f rowacc0 = {}, rowacc1 = {};
  for (int ntb = 0; ntb < 8; ++ntb) {
    const int nt = ntb * 8 + warp;
    v8f as0 = {}, ao0 = {}, au0 = {}, as1 = {}, ao1 = {}, au1 = {};
    for (int kt = 0; kt < DD / 32; ++kt) {
      const int s6 = kt * 6;
      v16bf bs = frag_b(Wfs, nt, kt, DD / 32, lane);
      v16bf bo = frag_b(Wfo, nt, kt, DD / 32, lane);
      v16bf bu = frag_b(Wfu, nt, kt, DD / 32, lane);
      as0 = wmma_bf16(frag_slot(A, s6 + 0, lane), bs, as0);
      as1 = wmma_bf16(frag_slot(A, s6 + 1, lane), bs, as1);
      ao0 = wmma_bf16(frag_slot(A, s6 + 2, lane), bo, ao0);
      ao1 = wmma_bf16(frag_slot(A, s6 + 3, lane), bo, ao1);
      au0 = wmma_bf16(frag_slot(A, s6 + 4, lane), bu, au0);
      au1 = wmma_bf16(frag_slot(A, s6 + 5, lane), bu, au1);
    }
    const int n = nt * 16 + m;
    const float bs = ws_b[n], bo = wo_b[n], bu = wu_b[n], wv = w_w[n];
#pragma unroll
    for (int r = 0; r < 8; ++r) {
      rowacc0[r] += (as0[r] + bs) * (ao0[r] + bo) * (au0[r] + bu) * wv;
      rowacc1[r] += (as1[r] + bs) * (ao1[r] + bo) * (au1[r] + bu) * wv;
    }
  }
  // reduce over the 16 lanes of each half (columns of this wave's tiles)
#pragma unroll
  for (int off = 8; off > 0; off >>= 1) {
#pragma unroll
    for (int r = 0; r < 8; ++r) {
      rowacc0[r] += __shfl_xor(rowacc0[r], off, 16);
      rowacc1[r] += __shfl_xor(rowacc1[r], off, 16);
    }
  }
  if (m == 0) {
    const int hh = lane >> 4;
#pragma unroll
    for (int r = 0; r < 8; ++r) {
      red[warp * 32 + hh * 8 + r] = rowacc0[r];
      red[warp * 32 + 16 + hh * 8 + r] = rowacc1[r];
    }
  }
  __syncthreads();
  if (threadIdx.x < 32) {
    float s = 0.f;
#pragma unroll
    for (int w = 0; w < 8; ++w) s += red[w * 32 + threadIdx.x];
    coeff[(size_t)b * RR + pg * 32 + threadIdx.x] = s + w_b[0];
  }
}

// ---------------- A build: zero, scatter-add, sigmoid/mask/row-normalize ----
__global__ void k_zero(float* __restrict__ p, size_t n) {
  size_t i = (size_t)blockIdx.x * blockDim.x + threadIdx.x;
  if (i < n) p[i] = 0.f;
}

__global__ void k_scatter(const int* __restrict__ pairs,
                          const float* __restrict__ coeff,
                          float* __restrict__ A) {
  int i = blockIdx.x * blockDim.x + threadIdx.x;
  if (i >= BB * RR) return;
  int b = i / RR;
  int p0 = pairs[(size_t)i * 2 + 0];
  int p1 = pairs[(size_t)i * 2 + 1];
  atomicAdd(&A[((size_t)b * NN + p0) * NN + p1], coeff[i]);
}

__global__ __launch_bounds__(256) void k_norm(const float* __restrict__ A,
                                              __bf16* __restrict__ Abf,
                                              __bf16* __restrict__ ATbf) {
  __shared__ float red[256];
  const int bi = blockIdx.x;         // b*256 + row i
  const int b = bi >> 8, i = bi & 255, j = threadIdx.x;
  float x = A[((size_t)b * NN + i) * NN + j];
  float sg = (j == i) ? 0.f : 1.f / (1.f + __expf(-x));
  red[j] = sg;
  __syncthreads();
  for (int s = 128; s > 0; s >>= 1) {
    if (j < s) red[j] += red[j + s];
    __syncthreads();
  }
  float v = sg / red[0];
  Abf[((size_t)b * NN + i) * NN + j] = (__bf16)v;
  ATbf[((size_t)b * NN + j) * NN + i] = (__bf16)v;
}

// ---------------- feat = relu(obj @ t3_w + b), stored in B-fragment order ---
__global__ __launch_bounds__(256) void k_feat(const float* __restrict__ obj,
                                              const __bf16* __restrict__ t3f,
                                              const float* __restrict__ t3_b,
                                              __bf16* __restrict__ featf) {
  const int lane = threadIdx.x & 31;
  const int wid = blockIdx.x * 8 + (threadIdx.x >> 5);
  const int nt = wid & 31;          // N=512 -> 32 ntiles
  const int mt = wid >> 5;          // B*256/16 = 512 mtiles
  const int b = mt >> 4;
  const int mloc0 = (mt & 15) * 16;
  const int m = lane & 15;
  const float* arow = obj + ((size_t)b * NN + mloc0 + m) * DD;
  v8f acc = {};
  for (int kt = 0; kt < DD / 32; ++kt)
    acc = wmma_bf16(frag_a_f32(arow, kt * 32, lane),
                    frag_b(t3f, nt, kt, DD / 32, lane), acc);
  const int hh = lane >> 4;
  const int n = nt * 16 + m;
  const float bias = t3_b[n];
  __bf16* dst = featf + (size_t)b * (256 * 512);
#pragma unroll
  for (int r = 0; r < 8; ++r) {
    int mo = mloc0 + hh * 8 + r;    // consumer k index (K=256, Kt=8)
    float v = acc[r] + bias;
    v = v > 0.f ? v : 0.f;
    int ktc = mo >> 5, kin = mo & 31;
    int lanec = ((kin >> 3) & 1) * 16 + (n & 15);
    dst[(((size_t)(n >> 4) * 8 + ktc) << 9) + lanec * 16 + ((kin >= 16 ? 8 : 0) + (kin & 7))] = (__bf16)v;
  }
}

// ---------------- ctx = [A @ feat, A^T @ feat] (bf16, row-major) ------------
__global__ __launch_bounds__(256) void k_ctx(const __bf16* __restrict__ Abf,
                                             const __bf16* __restrict__ ATbf,
                                             const __bf16* __restrict__ featf,
                                             __bf16* __restrict__ ctx) {
  const int lane = threadIdx.x & 31;
  int wid = blockIdx.x * 8 + (threadIdx.x >> 5);
  const int nt = wid & 31;  wid >>= 5;
  const int mt = wid & 15;  wid >>= 4;
  const int dir = wid & 1;
  const int b = wid >> 1;
  const int m = lane & 15;
  const __bf16* Am = (dir ? ATbf : Abf) + (size_t)b * (NN * NN);
  const __bf16* arow = Am + (size_t)(mt * 16 + m) * NN;
  const __bf16* Bf = featf + (size_t)b * (256 * 512);
  v8f acc = {};
  for (int kt = 0; kt < 8; ++kt)
    acc = wmma_bf16(frag_a_bf16(arow, kt * 32, lane),
                    frag_b(Bf, nt, kt, 8, lane), acc);
  const int hh = lane >> 4;
  const int n = dir * 512 + nt * 16 + m;
  __bf16* dst = ctx + (size_t)b * (256 * 1024);
#pragma unroll
  for (int r = 0; r < 8; ++r)
    dst[(size_t)(mt * 16 + hh * 8 + r) * 1024 + n] = (__bf16)acc[r];
}

// ------- fused tail: h = ctx@tr1 + b -> LN -> relu -> @tr2 + b + obj, relu --
// grid = BB*4: one independent 64-row chunk per block.
__global__ __launch_bounds__(256) void k_trans(
    const __bf16* __restrict__ ctx, const __bf16* __restrict__ tr1f,
    const float* __restrict__ tr1_b, const float* __restrict__ ln_g,
    const float* __restrict__ ln_b, const __bf16* __restrict__ tr2f,
    const float* __restrict__ tr2_b, const float* __restrict__ obj,
    float* __restrict__ out) {
  __shared__ float hbuf[64 * 256];   // 64 KB chunk of h (64 rows)
  const int b = blockIdx.x >> 2;
  const int c = blockIdx.x & 3;
  const int lane = threadIdx.x & 31;
  const int warp = threadIdx.x >> 5;
  const int m = lane & 15, hh = lane >> 4;
  const __bf16* ctxb = ctx + (size_t)b * (256 * 1024);
  // GEMM1: 64 rows x 256, K=1024  (4 mtiles x 16 ntiles, 8 waves)
  for (int t = warp; t < 64; t += 8) {
    const int mt = t >> 4, nt = t & 15;
    const __bf16* arow = ctxb + (size_t)(c * 64 + mt * 16 + m) * 1024;
    v8f acc = {};
    for (int kt = 0; kt < 32; ++kt)
      acc = wmma_bf16(frag_a_bf16(arow, kt * 32, lane),
                      frag_b(tr1f, nt, kt, 32, lane), acc);
    const int n = nt * 16 + m;
    const float bias = tr1_b[n];
#pragma unroll
    for (int r = 0; r < 8; ++r)
      hbuf[(mt * 16 + hh * 8 + r) * 256 + n] = acc[r] + bias;
  }
  __syncthreads();
  // LayerNorm + ReLU per row
  if (threadIdx.x < 64) {
    float* rowp = hbuf + threadIdx.x * 256;
    float mu = 0.f;
    for (int j = 0; j < 256; ++j) mu += rowp[j];
    mu *= (1.f / 256.f);
    float var = 0.f;
    for (int j = 0; j < 256; ++j) { float d = rowp[j] - mu; var += d * d; }
    var *= (1.f / 256.f);
    float inv = rsqrtf(var + LN_EPS);
    for (int j = 0; j < 256; ++j) {
      float v = (rowp[j] - mu) * inv * ln_g[j] + ln_b[j];
      rowp[j] = v > 0.f ? v : 0.f;
    }
  }
  __syncthreads();
  // GEMM2: 64 rows x 1024, K=256  (4 mtiles x 64 ntiles, 8 waves)
  for (int t = warp; t < 256; t += 8) {
    const int mt = t >> 6, nt = t & 63;
    const float* arow = hbuf + (size_t)(mt * 16 + m) * 256;
    v8f acc = {};
    for (int kt = 0; kt < 8; ++kt)
      acc = wmma_bf16(frag_a_f32(arow, kt * 32, lane),
                      frag_b(tr2f, nt, kt, 8, lane), acc);
    const int n = nt * 16 + m;
    const float bias = tr2_b[n];
#pragma unroll
    for (int r = 0; r < 8; ++r) {
      int ml = c * 64 + mt * 16 + hh * 8 + r;
      size_t o = ((size_t)b * NN + ml) * DD + n;
      float v = acc[r] + bias + obj[o];
      out[o] = v > 0.f ? v : 0.f;
    }
  }
}

extern "C" void kernel_launch(void* const* d_in, const int* in_sizes, int n_in,
                              void* d_out, int out_size, void* d_ws, size_t ws_size,
                              hipStream_t stream) {
  const float* obj  = (const float*)d_in[0];
  const float* uni  = (const float*)d_in[1];
  const float* ws_w = (const float*)d_in[2];
  const float* ws_b = (const float*)d_in[3];
  const float* wo_w = (const float*)d_in[4];
  const float* wo_b = (const float*)d_in[5];
  const float* wu_w = (const float*)d_in[6];
  const float* wu_b = (const float*)d_in[7];
  const float* w_w  = (const float*)d_in[8];
  const float* w_b  = (const float*)d_in[9];
  const float* t3_w = (const float*)d_in[10];
  const float* t3_b = (const float*)d_in[11];
  const float* tr1w = (const float*)d_in[12];
  const float* tr1b = (const float*)d_in[13];
  const float* ln_g = (const float*)d_in[14];
  const float* ln_b = (const float*)d_in[15];
  const float* tr2w = (const float*)d_in[16];
  const float* tr2b = (const float*)d_in[17];
  const int*  pairs = (const int*)d_in[18];
  float* out = (float*)d_out;

  char* ws = (char*)d_ws;
  size_t off = 0;
  auto alloc = [&](size_t bytes) {
    char* p = ws + off;
    off += (bytes + 255) & ~(size_t)255;
    return p;
  };
  __bf16* Wfs  = (__bf16*)alloc((size_t)DD * DD * 2);
  __bf16* Wfo  = (__bf16*)alloc((size_t)DD * DD * 2);
  __bf16* Wfu  = (__bf16*)alloc((size_t)DD * DD * 2);
  __bf16* t3f  = (__bf16*)alloc((size_t)DD * 512 * 2);
  __bf16* tr1f = (__bf16*)alloc((size_t)DD * 256 * 2);
  __bf16* tr2f = (__bf16*)alloc((size_t)256 * DD * 2);
  float*  coeff= (float*)alloc((size_t)BB * RR * 4);
  float*  Afp  = (float*)alloc((size_t)BB * NN * NN * 4);
  __bf16* Abf  = (__bf16*)alloc((size_t)BB * NN * NN * 2);
  __bf16* ATbf = (__bf16*)alloc((size_t)BB * NN * NN * 2);
  __bf16* featf= (__bf16*)alloc((size_t)BB * 256 * 512 * 2);
  __bf16* ctxb = (__bf16*)alloc((size_t)BB * 256 * 1024 * 2);

  // 1) pre-swizzle weights into WMMA B-fragment order (bf16)
  k_swizzle_w<<<(DD * DD) / 256, 256, 0, stream>>>(ws_w, Wfs, DD, DD);
  k_swizzle_w<<<(DD * DD) / 256, 256, 0, stream>>>(wo_w, Wfo, DD, DD);
  k_swizzle_w<<<(DD * DD) / 256, 256, 0, stream>>>(wu_w, Wfu, DD, DD);
  k_swizzle_w<<<(DD * 512) / 256, 256, 0, stream>>>(t3_w, t3f, DD, 512);
  k_swizzle_w<<<(DD * 256) / 256, 256, 0, stream>>>(tr1w, tr1f, DD, 256);
  k_swizzle_w<<<(256 * DD) / 256, 256, 0, stream>>>(tr2w, tr2f, 256, DD);

  // 2) fused gather-GEMM x3 + triple product -> coeff
  //    32 pairs/block, 192KB interleaved-slot LDS staging, B shared 2x
  const size_t coeff_lds = 6 * 32768 * sizeof(__bf16) + 8 * 32 * sizeof(float);
  k_coeff<<<BB * (RR / 32), 256, coeff_lds, stream>>>(
      obj, uni, Wfs, Wfo, Wfu, ws_b, wo_b, wu_b, w_w, w_b, pairs, coeff);

  // 3) attention matrix: zero, scatter-add, sigmoid/mask/row-normalize
  k_zero<<<(BB * NN * NN) / 256, 256, 0, stream>>>(Afp, (size_t)BB * NN * NN);
  k_scatter<<<(BB * RR) / 256, 256, 0, stream>>>(pairs, coeff, Afp);
  k_norm<<<BB * NN, 256, 0, stream>>>(Afp, Abf, ATbf);

  // 4) feat = relu(obj @ t3_w + b) stored fragment-ready
  k_feat<<<((BB * 256 / 16) * 32) / 8, 256, 0, stream>>>(obj, t3f, t3_b, featf);

  // 5) ctx = [A@feat, A^T@feat]
  k_ctx<<<(BB * 2 * 16 * 32) / 8, 256, 0, stream>>>(Abf, ATbf, featf, ctxb);

  // 6) fused tr1 -> LayerNorm -> relu -> tr2 -> +obj -> relu (per 64-row chunk)
  k_trans<<<BB * 4, 256, 0, stream>>>(ctxb, tr1f, tr1b, ln_g, ln_b, tr2f, tr2b,
                                      obj, out);
  (void)in_sizes; (void)n_in; (void)out_size; (void)ws_size;
}
